// CapsuleNet_73641509257426
// MI455X (gfx1250) — matrix operand order repaired and checked by reading, not compiled, not run on hardware
//
#include <hip/hip_runtime.h>
#include <hip/hip_bf16.h>

// ---------------------------------------------------------------------------
// CapsuleNet forward for MI455X (gfx1250, wave32, WMMA).
// Heavy GEMMs use v_wmma_f32_16x16x32_f16 (f16 in, f32 accumulate).
// gemm_pc is K-split 4 ways for occupancy (576 -> 2304 waves) with a
// deterministic partial-sum reduction (no fp atomics).
// ---------------------------------------------------------------------------

typedef __attribute__((ext_vector_type(16))) _Float16 v16h;
typedef __attribute__((ext_vector_type(8)))  _Float16 v8h;
typedef __attribute__((ext_vector_type(8)))  float    v8f;

union AFrag { v16h v; v8h h[2]; };

static __device__ __forceinline__ v8f wmma16(v16h a, v16h b, v8f c) {
  // (neg_a, A, neg_b, B, c_mod, C, reuse_a, reuse_b)
  return __builtin_amdgcn_wmma_f32_16x16x32_f16(false, a, false, b, (short)0, c,
                                                false, false);
}

#define BATCH   256
#define N1      102400     // BATCH * 20 * 20  (conv1 output positions)
#define K1P     96         // 81 padded to multiple of 32
#define N2      9216       // BATCH * 36       (pc-conv output positions)
#define K2      20736      // 256 * 81
#define NPRIM   1152
#define KSPLIT  4          // K-split factor for gemm_pc (81 taps -> 21/21/21/18)

// ---------------------------------------------------------------------------
// Packing / im2col kernels
// ---------------------------------------------------------------------------
__global__ void pack_c1w(const float* __restrict__ w, _Float16* __restrict__ o) {
  int id = blockIdx.x * blockDim.x + threadIdx.x;       // 256*96
  if (id >= 256 * K1P) return;
  int oc = id / K1P, k = id % K1P;
  o[id] = (k < 81) ? (_Float16)w[oc * 81 + k] : (_Float16)0.f;
}

__global__ void im2col1(const float* __restrict__ x, _Float16* __restrict__ o) {
  long id = (long)blockIdx.x * blockDim.x + threadIdx.x; // N1*96
  if (id >= (long)N1 * K1P) return;
  int n = (int)(id / K1P), k = (int)(id % K1P);
  _Float16 v = (_Float16)0.f;
  if (k < 81) {
    int ky = k / 9, kx = k % 9;
    int b = n / 400, s = n % 400, oy = s / 20, ox = s % 20;
    v = (_Float16)x[b * 784 + (oy + ky) * 28 + (ox + kx)];
  }
  o[id] = v;
}

// pc_w [oc][ic][ky][kx] f32 -> Wp [oc][ (ky*9+kx)*256 + ic ] f16
__global__ void pack_pcw(const float* __restrict__ w, _Float16* __restrict__ o) {
  long id = (long)blockIdx.x * blockDim.x + threadIdx.x; // 256*256*81
  if (id >= (long)256 * K2) return;
  int oc = (int)(id / K2); int r = (int)(id % K2);
  int kyx = r / 256, ic = r % 256;
  o[id] = (_Float16)w[((long)oc * 256 + ic) * 81 + kyx];
}

// ---------------------------------------------------------------------------
// GEMM 1: conv1 as explicit-im2col GEMM.  C[m][n] = sum_k A[m][k]*Bt[n][k]
// A = conv1 weights f16 [256][96], Bt = im2col [N1][96].
// Epilogue: relu(acc + bias), store y channels-last f16 [n][256].
// One wave computes a 64x64 tile (4x4 of 16x16 WMMA accumulators).
// ---------------------------------------------------------------------------
__global__ __launch_bounds__(32)
void gemm_conv1(const _Float16* __restrict__ A, const _Float16* __restrict__ Bt,
                const float* __restrict__ bias, _Float16* __restrict__ Y) {
  const int lane = threadIdx.x;
  const int lo = lane & 15, hi = lane >> 4;
  const int m0 = blockIdx.x * 64;       // 0..3
  const int n0 = blockIdx.y * 64;       // 0..1599

  const v8f zero = {0.f,0.f,0.f,0.f,0.f,0.f,0.f,0.f};
  v8f acc[4][4];
#pragma unroll
  for (int i = 0; i < 4; i++)
#pragma unroll
    for (int j = 0; j < 4; j++) acc[i][j] = zero;

  long aRow[4], bRow[4];
#pragma unroll
  for (int i = 0; i < 4; i++) aRow[i] = (long)(m0 + 16*i + lo) * K1P + hi * 16;
#pragma unroll
  for (int j = 0; j < 4; j++) bRow[j] = (long)(n0 + 16*j + lo) * K1P + hi * 16;

  for (int k0 = 0; k0 < K1P; k0 += 32) {
    AFrag a[4]; v16h b[4];
#pragma unroll
    for (int i = 0; i < 4; i++) {
      a[i].h[0] = *(const v8h*)(A + aRow[i] + k0);       // K = k0+hi*16  .. +7
      a[i].h[1] = *(const v8h*)(A + aRow[i] + k0 + 8);   // K = k0+hi*16+8.. +15
    }
#pragma unroll
    for (int j = 0; j < 4; j++)
      b[j] = *(const v16h*)(Bt + bRow[j] + k0);          // K = k0+hi*16  .. +15
#pragma unroll
    for (int i = 0; i < 4; i++)
#pragma unroll
      for (int j = 0; j < 4; j++) acc[i][j] = wmma16(a[i].v, b[j], acc[i][j]);
  }

#pragma unroll
  for (int i = 0; i < 4; i++) {
    int mBase = m0 + 16*i + 8*hi;       // C/D layout: VGPR v -> M = v + 8*hi
    float bv[8];
#pragma unroll
    for (int v = 0; v < 8; v++) bv[v] = bias[mBase + v];
#pragma unroll
    for (int j = 0; j < 4; j++) {
      int n = n0 + 16*j + lo;
      v8h o;
#pragma unroll
      for (int v = 0; v < 8; v++)
        o[v] = (_Float16)fmaxf(acc[i][j][v] + bv[v], 0.f);
      *(v8h*)(Y + (long)n * 256 + mBase) = o;
    }
  }
}

// ---------------------------------------------------------------------------
// GEMM 2: primary-capsule conv, implicit im2col (gathers from channels-last y).
// M=oc(256), N=b*36+s(9216), K=(ky,kx,ic)(20736), K-split over blockIdx.z.
// Writes fp32 partials Ppart[z][n][m] (contiguous v8f stores).
// ---------------------------------------------------------------------------
__global__ __launch_bounds__(32)
void gemm_pc(const _Float16* __restrict__ Wp, const _Float16* __restrict__ Y,
             float* __restrict__ Ppart) {
  const int lane = threadIdx.x;
  const int lo = lane & 15, hi = lane >> 4;
  const int m0 = blockIdx.x * 64;       // 0..3
  const int n0 = blockIdx.y * 64;       // 0..143
  const int z  = blockIdx.z;            // 0..KSPLIT-1
  const int kyx0 = z * 21;
  const int kyx1 = (kyx0 + 21 < 81) ? kyx0 + 21 : 81;

  const v8f zero = {0.f,0.f,0.f,0.f,0.f,0.f,0.f,0.f};
  v8f acc[4][4];
#pragma unroll
  for (int i = 0; i < 4; i++)
#pragma unroll
    for (int j = 0; j < 4; j++) acc[i][j] = zero;

  long aRow[4];
#pragma unroll
  for (int i = 0; i < 4; i++) aRow[i] = (long)(m0 + 16*i + lo) * K2 + hi * 16;

  long bBase[4];
#pragma unroll
  for (int j = 0; j < 4; j++) {
    int n = n0 + 16*j + lo;
    int b = n / 36, s = n % 36, oy = s / 6, ox = s % 6;
    bBase[j] = ((long)(b * 20 + 2 * oy) * 20 + 2 * ox) * 256 + hi * 16;
  }

  for (int kyx = kyx0; kyx < kyx1; kyx++) {
    const int ky = kyx / 9, kx = kyx % 9;
    const long yOff = (long)(ky * 20 + kx) * 256;
    const long aK = (long)kyx * 256;
    for (int icb = 0; icb < 256; icb += 32) {
      AFrag a[4]; v16h b[4];
#pragma unroll
      for (int i = 0; i < 4; i++) {
        a[i].h[0] = *(const v8h*)(Wp + aRow[i] + aK + icb);
        a[i].h[1] = *(const v8h*)(Wp + aRow[i] + aK + icb + 8);
      }
#pragma unroll
      for (int j = 0; j < 4; j++)
        b[j] = *(const v16h*)(Y + bBase[j] + yOff + icb);
#pragma unroll
      for (int i = 0; i < 4; i++)
#pragma unroll
        for (int j = 0; j < 4; j++) acc[i][j] = wmma16(a[i].v, b[j], acc[i][j]);
    }
  }

  float* base = Ppart + (long)z * N2 * 256;
#pragma unroll
  for (int i = 0; i < 4; i++) {
    int mBase = m0 + 16*i + 8*hi;       // VGPR v -> M = v + 8*hi : 8 contiguous m
#pragma unroll
    for (int j = 0; j < 4; j++) {
      int n = n0 + 16*j + lo;
      *(v8f*)(base + (long)n * 256 + mBase) = acc[i][j];
    }
  }
}

// Reduce K-split partials, add conv bias, scatter into capsule layout
// PC[b][g*36+s][d] with oc = d*32+g, n = b*36+s.
__global__ void pc_reduce(const float* __restrict__ Ppart,
                          const float* __restrict__ bias,
                          float* __restrict__ PC) {
  int id = blockIdx.x * blockDim.x + threadIdx.x;   // N2*256
  if (id >= N2 * 256) return;
  int n = id >> 8, m = id & 255;
  float acc = bias[m];
#pragma unroll
  for (int zz = 0; zz < KSPLIT; zz++)
    acc += Ppart[(long)zz * N2 * 256 + id];
  int b = n / 36, s = n % 36;
  int d = m >> 5, g = m & 31;
  PC[((long)b * NPRIM + g * 36 + s) * 8 + d] = acc;
}

// ---------------------------------------------------------------------------
// squash over the 8 primary-capsule dims (in place)
// ---------------------------------------------------------------------------
__global__ void squash_u(float* __restrict__ U) {
  int id = blockIdx.x * blockDim.x + threadIdx.x;   // BATCH*NPRIM
  if (id >= BATCH * NPRIM) return;
  float* p = U + (long)id * 8;
  float x[8], sq = 0.f;
#pragma unroll
  for (int i = 0; i < 8; i++) { x[i] = p[i]; sq = fmaf(x[i], x[i], sq); }
  float sc = (sq / (1.f + sq)) / sqrtf(sq + 1e-7f);
#pragma unroll
  for (int i = 0; i < 8; i++) p[i] = x[i] * sc;
}

// ---------------------------------------------------------------------------
// u_hat = einsum('ncdp,bnp->bncd'); store f16 [b][n][c][d]
// ---------------------------------------------------------------------------
__global__ void uhat_kernel(const float* __restrict__ Wr, const float* __restrict__ U,
                            _Float16* __restrict__ WU) {
  long id = (long)blockIdx.x * blockDim.x + threadIdx.x; // BATCH*NPRIM*160
  if (id >= (long)BATCH * NPRIM * 160) return;
  int b  = (int)(id / ((long)NPRIM * 160));
  int r  = (int)(id % ((long)NPRIM * 160));
  int n  = r / 160, cd = r % 160;
  const float* w = Wr + ((long)n * 160 + cd) * 8;
  const float* u = U + ((long)b * NPRIM + n) * 8;
  float acc = 0.f;
#pragma unroll
  for (int p = 0; p < 8; p++) acc = fmaf(w[p], u[p], acc);
  WU[id] = (_Float16)acc;
}

// ---------------------------------------------------------------------------
// Dynamic routing: one 160-thread block per batch element.
// b_ij rows live in registers (each thread owns n = t, t+160, ...);
// c_ij in LDS (46 KB).  Writes clf -> d_out, masked v -> H0.
// ---------------------------------------------------------------------------
__global__ __launch_bounds__(160)
void routing_kernel(const _Float16* __restrict__ WU, const int* __restrict__ labels,
                    float* __restrict__ clf, float* __restrict__ H0) {
  __shared__ float cij[NPRIM * 10];
  __shared__ float sv[160];
  __shared__ float vv[160];

  const int t = threadIdx.x, bb = blockIdx.x;
  const int c = t >> 4, d = t & 15;
  const _Float16* wu = WU + (long)bb * NPRIM * 160;

  int rows[8], nrows = 0;
  for (int n = t; n < NPRIM; n += 160) rows[nrows++] = n;
  float bj[8][10];
  for (int r = 0; r < nrows; r++)
    for (int cc = 0; cc < 10; cc++) bj[r][cc] = 0.f;

  for (int iter = 0; iter < 3; iter++) {
    // c_ij = softmax over classes of b_ij
    for (int r = 0; r < nrows; r++) {
      float mx = bj[r][0];
      for (int cc = 1; cc < 10; cc++) mx = fmaxf(mx, bj[r][cc]);
      float e[10], s = 0.f;
      for (int cc = 0; cc < 10; cc++) { e[cc] = expf(bj[r][cc] - mx); s += e[cc]; }
      float inv = 1.f / s;
      for (int cc = 0; cc < 10; cc++) cij[rows[r] * 10 + cc] = e[cc] * inv;
    }
    __syncthreads();

    // s_sum[c][d] = sum_n c_ij[n][c] * WU[n][c][d]
    float acc = 0.f;
    for (int n = 0; n < NPRIM; n++)
      acc = fmaf(cij[n * 10 + c], (float)wu[(long)n * 160 + t], acc);
    sv[t] = acc;
    __syncthreads();

    float sq = 0.f;
    for (int dd = 0; dd < 16; dd++) { float x = sv[c * 16 + dd]; sq = fmaf(x, x, sq); }
    float vval = (sq / (1.f + sq)) * acc / sqrtf(sq + 1e-7f);
    vv[t] = vval;
    __syncthreads();

    if (iter < 2) {
      // b_ij += c_ij * dot(v, WU)  (agreement uses s = c*WU, as in reference)
      for (int r = 0; r < nrows; r++) {
        int n = rows[r];
        for (int cc = 0; cc < 10; cc++) {
          float dot = 0.f;
          for (int dd = 0; dd < 16; dd++)
            dot = fmaf(vv[cc * 16 + dd], (float)wu[(long)n * 160 + cc * 16 + dd], dot);
          bj[r][cc] += cij[n * 10 + cc] * dot;
        }
      }
      __syncthreads();
    } else {
      if (d == 0) {
        float s2 = 0.f;
        for (int dd = 0; dd < 16; dd++) { float x = vv[c * 16 + dd]; s2 = fmaf(x, x, s2); }
        clf[bb * 10 + c] = sqrtf(s2);
      }
      int lab = labels[bb];
      H0[bb * 160 + t] = (c == lab) ? vval : 0.f;
    }
  }
}

// ---------------------------------------------------------------------------
// Decoder FC layers (tiny; VALU fp32).  act: 0 = relu, 1 = sigmoid
// ---------------------------------------------------------------------------
__global__ void fc_kernel(const float* __restrict__ X, const float* __restrict__ W,
                          const float* __restrict__ bias, float* __restrict__ O,
                          int K, int N, int act) {
  int id = blockIdx.x * blockDim.x + threadIdx.x;
  if (id >= BATCH * N) return;
  int b = id / N, j = id % N;
  const float* x = X + (long)b * K;
  float acc = bias[j];
  for (int k = 0; k < K; k++) acc = fmaf(x[k], W[(long)k * N + j], acc);
  O[id] = (act == 0) ? fmaxf(acc, 0.f) : 1.f / (1.f + expf(-acc));
}

// ---------------------------------------------------------------------------
extern "C" void kernel_launch(void* const* d_in, const int* in_sizes, int n_in,
                              void* d_out, int out_size, void* d_ws, size_t ws_size,
                              hipStream_t stream) {
  const float* x      = (const float*)d_in[0];
  const int*   labels = (const int*)d_in[1];
  const float* c1w    = (const float*)d_in[2];
  const float* c1b    = (const float*)d_in[3];
  const float* pcw    = (const float*)d_in[4];
  const float* pcb    = (const float*)d_in[5];
  const float* rw     = (const float*)d_in[6];
  const float* w1     = (const float*)d_in[7];
  const float* b1     = (const float*)d_in[8];
  const float* w2     = (const float*)d_in[9];
  const float* b2     = (const float*)d_in[10];
  const float* w3     = (const float*)d_in[11];
  const float* b3     = (const float*)d_in[12];
  float* out = (float*)d_out;            // clf [256*10] ++ rec [256*784]

  char* ws = (char*)d_ws;
  size_t off = 0;
  auto take = [&](size_t bytes) -> void* {
    void* p = ws + off;
    off += (bytes + 255) & ~(size_t)255;
    return p;
  };
  _Float16* Wc1   = (_Float16*)take((size_t)256 * K1P * 2);            //  48 KB
  _Float16* Bt1   = (_Float16*)take((size_t)N1 * K1P * 2);             //  19 MB
  _Float16* Yh    = (_Float16*)take((size_t)N1 * 256 * 2);             //  52 MB
  _Float16* Wp    = (_Float16*)take((size_t)256 * K2 * 2);             //  10 MB
  float*    Ppart = (float*)take((size_t)KSPLIT * N2 * 256 * 4);       //  38 MB
  float*    U     = (float*)take((size_t)BATCH * NPRIM * 8 * 4);       //   9 MB
  _Float16* WU    = (_Float16*)take((size_t)BATCH * NPRIM * 160 * 2);  //  94 MB
  float*    H0    = (float*)take((size_t)BATCH * 160 * 4);
  float*    H1    = (float*)take((size_t)BATCH * 512 * 4);
  float*    H2    = (float*)take((size_t)BATCH * 1024 * 4);
  (void)ws_size; (void)n_in; (void)in_sizes; (void)out_size;

  // conv1 as GEMM
  pack_c1w<<<(256 * K1P + 255) / 256, 256, 0, stream>>>(c1w, Wc1);
  im2col1<<<(int)(((long)N1 * K1P + 255) / 256), 256, 0, stream>>>(x, Bt1);
  gemm_conv1<<<dim3(4, N1 / 64), 32, 0, stream>>>(Wc1, Bt1, c1b, Yh);

  // primary-capsule conv as implicit GEMM (K-split) + deterministic reduce
  pack_pcw<<<(int)(((long)256 * K2 + 255) / 256), 256, 0, stream>>>(pcw, Wp);
  gemm_pc<<<dim3(4, N2 / 64, KSPLIT), 32, 0, stream>>>(Wp, Yh, Ppart);
  pc_reduce<<<(N2 * 256 + 255) / 256, 256, 0, stream>>>(Ppart, pcb, U);

  // squash -> u, u_hat, routing
  squash_u<<<(BATCH * NPRIM + 255) / 256, 256, 0, stream>>>(U);
  uhat_kernel<<<(int)(((long)BATCH * NPRIM * 160 + 255) / 256), 256, 0, stream>>>(rw, U, WU);
  routing_kernel<<<BATCH, 160, 0, stream>>>(WU, labels, out, H0);

  // decoder
  fc_kernel<<<(BATCH * 512 + 255) / 256, 256, 0, stream>>>(H0, w1, b1, H1, 160, 512, 0);
  fc_kernel<<<(BATCH * 1024 + 255) / 256, 256, 0, stream>>>(H1, w2, b2, H2, 512, 1024, 0);
  fc_kernel<<<(BATCH * 784 + 255) / 256, 256, 0, stream>>>(H2, w3, b3, out + BATCH * 10,
                                                           1024, 784, 1);
}
